// Linear_53515292508495
// MI455X (gfx1250) — matrix operand order repaired and checked by reading, not compiled, not run on hardware
//
#include <hip/hip_runtime.h>
#include <hip/hip_bf16.h>

// ---------------- problem constants ----------------
#define DIM   1024   // D
#define ODIM  1024   // O
#define NEXP  8      // E
#define RNK   16     // R
#define ER    128    // E*R
#define NTOK  4096   // B*S
#define SCAL  2.0f   // lora_alpha / r = 32/16

// ---------------- types ----------------
typedef __attribute__((ext_vector_type(4)))  __bf16 v4bf;
typedef __attribute__((ext_vector_type(8)))  __bf16 v8bf;
typedef __attribute__((ext_vector_type(16))) __bf16 v16bf;
typedef __attribute__((ext_vector_type(8)))  float  v8f;
typedef __attribute__((ext_vector_type(4)))  unsigned int u32x4;
typedef __attribute__((ext_vector_type(8)))  int i32x8;
typedef __attribute__((ext_vector_type(4)))  int i32x4;

#define LST 40   // LDS tile row stride in bf16 elements (80 bytes: 64B row + 16B TDM pad)

// round-to-nearest-even f32 -> bf16 (prep/epilogue paths only, not in GEMM hot loop)
__device__ __forceinline__ __bf16 tobf(float f) {
    union { float f; unsigned u; } a; a.f = f;
    unsigned r = a.u + 0x7FFFu + ((a.u >> 16) & 1u);
    unsigned short h = (unsigned short)(r >> 16);
    __bf16 o; __builtin_memcpy(&o, &h, 2); return o;
}

__device__ __forceinline__ unsigned lds_off(const void* p) {
    // generic LDS address: low 32 bits are the byte offset within the wave's LDS
    return (unsigned)(unsigned long long)p;
}

// ---- Tensor Data Mover: 2D bf16 tile (32 wide x 128 rows) global -> LDS ----
// pad_interval enc 3 = 16 DWORDs (one 64B row), pad_amount enc 3 = 4 DWORDs (16B)
// => LDS row stride 80B = LST bf16 elements.
__device__ __forceinline__ void tdm_tile_bf16(unsigned lds, const __bf16* g, int stride_elems) {
    unsigned long long ga = (unsigned long long)g;
    u32x4 g0 = { 1u,                                   // count=1, user descriptor
                 lds,                                  // lds_addr (bytes)
                 (unsigned)(ga & 0xFFFFFFFFu),         // global_addr[31:0]
                 ((unsigned)((ga >> 32) & 0x1FFFFFFull)) | 0x80000000u };  // ga[56:32] | type=2
    i32x8 g1 = { (1 << 16) | (1 << 20) | (3 << 22) | (3 << 25), // data_size=2B, pad_en, intv, amt
                 (int)(32u << 16),                     // tensor_dim0 = 32 (bits 79:48, low16 here)
                 (int)(128u << 16),                    // dim0 hi=0 | tensor_dim1=128 low16
                 (int)(32u << 16),                     // dim1 hi=0 | tile_dim0=32
                 128,                                  // tile_dim1=128, tile_dim2=0
                 stride_elems,                         // tensor_dim0_stride low32
                 0, 0 };
    i32x4 z4 = { 0, 0, 0, 0 };
    i32x8 z8 = { 0, 0, 0, 0, 0, 0, 0, 0 };
    __builtin_amdgcn_tensor_load_to_lds(g0, g1, z4, z4, z8, 0);
}

// A fragment (16x32, M x K), from row-major [M][LST] LDS tile.
// lane m=(l&15); lanes<16: K 0-7 & 16-23 ; lanes>=16: K 8-15 & 24-31.
__device__ __forceinline__ v16bf ldsA(const __bf16* base, int row, int lane) {
    int kk = (lane >> 4) * 8;
    const __bf16* p = base + (row + (lane & 15)) * LST + kk;
    v8bf lo = *(const v8bf*)(p);
    v8bf hi = *(const v8bf*)(p + 16);
    return __builtin_shufflevector(lo, hi, 0,1,2,3,4,5,6,7,8,9,10,11,12,13,14,15);
}

// B fragment (32x16, K x N) from row-major B^T = [N][LST] LDS tile.
// lane n=(l&15); lanes<16: K 0-7 & 16-23 ; lanes>=16: K 8-15 & 24-31.
__device__ __forceinline__ v16bf ldsB(const __bf16* base, int row, int lane) {
    int kk = (lane >> 4) * 8;
    const __bf16* p = base + (row + (lane & 15)) * LST + kk;
    v8bf lo = *(const v8bf*)(p);
    v8bf hi = *(const v8bf*)(p + 16);
    return __builtin_shufflevector(lo, hi, 0,1,2,3,4,5,6,7,8,9,10,11,12,13,14,15);
}

__device__ __forceinline__ v8f wmma_bf16(v16bf a, v16bf b, v8f c) {
    return __builtin_amdgcn_wmma_f32_16x16x32_bf16(false, a, false, b, (short)0, c, false, false);
}

// one 128x128-tile K-step: 4x2 fragments per wave (wave grid 2Mx4N)
__device__ __forceinline__ void compute_step(const __bf16* lx, const __bf16* lw,
                                             int wm, int wn, int lane, v8f acc[4][2]) {
    v16bf a[4], b[2];
#pragma unroll
    for (int mt = 0; mt < 4; ++mt) a[mt] = ldsA(lx, wm * 64 + mt * 16, lane);
#pragma unroll
    for (int nt = 0; nt < 2; ++nt) b[nt] = ldsB(lw, wn * 32 + nt * 16, lane);
#pragma unroll
    for (int mt = 0; mt < 4; ++mt)
#pragma unroll
        for (int nt = 0; nt < 2; ++nt)
            acc[mt][nt] = wmma_bf16(a[mt], b[nt], acc[mt][nt]);
}

// ---------------- prep kernels ----------------
// elementwise f32 -> bf16, 4 elements per thread
__global__ __launch_bounds__(256) void cvt_bf16_kernel(const float* __restrict__ s,
                                                       __bf16* __restrict__ d) {
    int i = blockIdx.x * 256 + threadIdx.x;
    float4 v = ((const float4*)s)[i];
    v4bf o = { tobf(v.x), tobf(v.y), tobf(v.z), tobf(v.w) };
    ((v4bf*)d)[i] = o;
}

// At[n][d] = lora_A[e][d][r], n = e*16+r   (row-major N x K for B fragments)
__global__ __launch_bounds__(256) void prep_At_kernel(const float* __restrict__ lora_A,
                                                      __bf16* __restrict__ At) {
    int idx = blockIdx.x * 256 + threadIdx.x;      // 128*1024 total
    int n = idx >> 10, d = idx & 1023;
    int e = n >> 4, r = n & 15;
    At[n * DIM + d] = tobf(lora_A[(e * DIM + d) * RNK + r]);
}

// Bt[o][c] = lora_B[e][r][o], c = e*16+r   (row-major N x K for B fragments)
__global__ __launch_bounds__(256) void prep_Bt_kernel(const float* __restrict__ lora_B,
                                                      __bf16* __restrict__ Bt) {
    int idx = blockIdx.x * 256 + threadIdx.x;      // 1024*128 total
    int o = idx >> 7, c = idx & 127;
    int e = c >> 4, r = c & 15;
    Bt[o * ER + c] = tobf(lora_B[(e * RNK + r) * ODIM + o]);
}

// ---------------- router: logits + top-2 softmax, SCALING folded ----------------
__global__ __launch_bounds__(256) void router_kernel(const float* __restrict__ x,
                                                     const float* __restrict__ rw,
                                                     const float* __restrict__ rb,
                                                     float* __restrict__ ew) {
    __shared__ float rwl[NEXP * DIM];
    __shared__ float rbl[NEXP];
    for (int i = threadIdx.x; i < NEXP * DIM; i += 256) rwl[i] = rw[i];
    if (threadIdx.x < NEXP) rbl[threadIdx.x] = rb[threadIdx.x];
    __syncthreads();

    int wid = threadIdx.x >> 5, lane = threadIdx.x & 31;
    int token = blockIdx.x * 8 + wid;
    const float* xr = x + token * DIM;

    float acc[NEXP];
#pragma unroll
    for (int e = 0; e < NEXP; ++e) acc[e] = 0.0f;

    for (int k = lane * 4; k < DIM; k += 128) {
        float4 xv = *(const float4*)(xr + k);
#pragma unroll
        for (int e = 0; e < NEXP; ++e) {
            float4 wv = *(const float4*)(&rwl[e * DIM + k]);
            acc[e] += xv.x * wv.x + xv.y * wv.y + xv.z * wv.z + xv.w * wv.w;
        }
    }
#pragma unroll
    for (int off = 16; off > 0; off >>= 1)
#pragma unroll
        for (int e = 0; e < NEXP; ++e) acc[e] += __shfl_xor(acc[e], off, 32);
#pragma unroll
    for (int e = 0; e < NEXP; ++e) acc[e] += rbl[e];

    float m1 = -3.4e38f, m2 = -3.4e38f;
#pragma unroll
    for (int e = 0; e < NEXP; ++e) {
        float v = acc[e];
        if (v > m1) { m2 = m1; m1 = v; } else if (v > m2) { m2 = v; }
    }
    float w[NEXP], sum = 0.0f;
#pragma unroll
    for (int e = 0; e < NEXP; ++e) {
        float v = (acc[e] >= m2) ? __expf(acc[e] - m1) : 0.0f;
        w[e] = v; sum += v;
    }
    float s = SCAL / sum;   // fold LoRA scaling into routing weight
    if (lane < NEXP) ew[token * NEXP + lane] = w[lane] * s;
}

// ---------------- GEMM1: wax = diag(ew) * (xb @ At^T), bf16 out (TDM pipelined) ----------------
__global__ __launch_bounds__(256) void moe_ax_kernel(const __bf16* __restrict__ xb,
                                                     const __bf16* __restrict__ At,
                                                     const float* __restrict__ ew,
                                                     __bf16* __restrict__ wax) {
    __shared__ __align__(128) __bf16 smem[2][2][128 * LST];  // [buf][x/w] : 40 KB
    __shared__ float lew[128 * NEXP];

    const int m0 = blockIdx.x * 128;
    const int t = threadIdx.x;
    const int lane = t & 31, wid = t >> 5;
    const int wm = wid >> 2, wn = wid & 3;

#pragma unroll
    for (int j = 0; j < 4; ++j) lew[t * 4 + j] = ew[m0 * NEXP + t * 4 + j];

    if (wid == 0) {   // prologue: DMA tile 0 into buffer 0
        tdm_tile_bf16(lds_off(&smem[0][0][0]), xb + m0 * DIM, DIM);
        tdm_tile_bf16(lds_off(&smem[0][1][0]), At, DIM);
    }

    v8f acc[4][2] = {};
    for (int it = 0; it < 32; ++it) {
        int cur = it & 1;
        if (wid == 0) {
            if (it + 1 < 32) {
                int k = (it + 1) * 32;
                tdm_tile_bf16(lds_off(&smem[cur ^ 1][0][0]), xb + m0 * DIM + k, DIM);
                tdm_tile_bf16(lds_off(&smem[cur ^ 1][1][0]), At + k, DIM);
                __builtin_amdgcn_s_wait_tensorcnt(2);   // tile `it` landed (in-order)
            } else {
                __builtin_amdgcn_s_wait_tensorcnt(0);
            }
        }
        __syncthreads();                                // broadcast tile ready
        compute_step(&smem[cur][0][0], &smem[cur][1][0], wm, wn, lane, acc);
        __syncthreads();                                // buffer reusable
    }

#pragma unroll
    for (int mt = 0; mt < 4; ++mt) {
        int rb = wm * 64 + mt * 16 + (lane >> 4) * 8;
#pragma unroll
        for (int nt = 0; nt < 2; ++nt) {
            int n = wn * 32 + nt * 16 + (lane & 15);
            int e = n >> 4;
#pragma unroll
            for (int i = 0; i < 8; ++i) {
                int rr = rb + i;
                wax[(m0 + rr) * ER + n] = tobf(acc[mt][nt][i] * lew[rr * NEXP + e]);
            }
        }
    }
}

// ---------------- GEMM2 (fused): out = xb@wb^T + b + wax@Bt^T (TDM pipelined) ----------------
// Unified 36-step K pipeline: steps 0-31 base GEMM (K=1024), steps 32-35 LoRA (K=128).
__global__ __launch_bounds__(256) void moe_main_kernel(const __bf16* __restrict__ xb,
                                                       const __bf16* __restrict__ wb,
                                                       const float* __restrict__ bias,
                                                       const __bf16* __restrict__ wax,
                                                       const __bf16* __restrict__ Bt,
                                                       float* __restrict__ out) {
    __shared__ __align__(128) __bf16 smem[2][2][128 * LST];  // 40 KB

    const int n0 = blockIdx.x * 128;
    const int m0 = blockIdx.y * 128;
    const int t = threadIdx.x;
    const int lane = t & 31, wid = t >> 5;
    const int wm = wid >> 2, wn = wid & 3;

    if (wid == 0) {   // prologue: step 0 sources
        tdm_tile_bf16(lds_off(&smem[0][0][0]), xb + m0 * DIM, DIM);
        tdm_tile_bf16(lds_off(&smem[0][1][0]), wb + n0 * DIM, DIM);
    }

    v8f acc[4][2] = {};
    for (int it = 0; it < 36; ++it) {
        int cur = it & 1;
        if (wid == 0) {
            int nx = it + 1;
            if (nx < 36) {
                const __bf16* sx; const __bf16* sw; int st;
                if (nx < 32) {
                    int k = nx * 32;
                    sx = xb + m0 * DIM + k; sw = wb + n0 * DIM + k; st = DIM;
                } else {
                    int k = (nx - 32) * 32;
                    sx = wax + m0 * ER + k; sw = Bt + n0 * ER + k; st = ER;
                }
                tdm_tile_bf16(lds_off(&smem[cur ^ 1][0][0]), sx, st);
                tdm_tile_bf16(lds_off(&smem[cur ^ 1][1][0]), sw, st);
                __builtin_amdgcn_s_wait_tensorcnt(2);   // tile `it` landed
            } else {
                __builtin_amdgcn_s_wait_tensorcnt(0);
            }
        }
        __syncthreads();
        compute_step(&smem[cur][0][0], &smem[cur][1][0], wm, wn, lane, acc);
        __syncthreads();
    }

    // epilogue: + bias, f32 store
#pragma unroll
    for (int mt = 0; mt < 4; ++mt) {
        int rbase = m0 + wm * 64 + mt * 16 + (lane >> 4) * 8;
#pragma unroll
        for (int nt = 0; nt < 2; ++nt) {
            int n = n0 + wn * 32 + nt * 16 + (lane & 15);
            float bv = bias[n];
#pragma unroll
            for (int i = 0; i < 8; ++i)
                out[(rbase + i) * ODIM + n] = acc[mt][nt][i] + bv;
        }
    }
}

// ---------------- launcher ----------------
extern "C" void kernel_launch(void* const* d_in, const int* in_sizes, int n_in,
                              void* d_out, int out_size, void* d_ws, size_t ws_size,
                              hipStream_t stream) {
    const float* x        = (const float*)d_in[0];
    const float* base_w   = (const float*)d_in[1];
    const float* base_b   = (const float*)d_in[2];
    const float* lora_A   = (const float*)d_in[3];
    const float* lora_B   = (const float*)d_in[4];
    const float* router_w = (const float*)d_in[5];
    const float* router_b = (const float*)d_in[6];
    float* out = (float*)d_out;

    char* ws = (char*)d_ws;
    __bf16* xb  = (__bf16*)(ws);                  // 4096*1024*2 = 8 MB
    __bf16* wb  = (__bf16*)(ws + 8388608);        // 1024*1024*2 = 2 MB
    __bf16* At  = (__bf16*)(ws + 10485760);       // 128*1024*2  = 256 KB
    __bf16* Bt  = (__bf16*)(ws + 10747904);       // 1024*128*2  = 256 KB
    float*  ew  = (float*) (ws + 11010048);       // 4096*8*4    = 128 KB
    __bf16* wax = (__bf16*)(ws + 11141120);       // 4096*128*2  = 1 MB

    cvt_bf16_kernel<<<4096, 256, 0, stream>>>(x, xb);        // 4M elems, 4/thread
    cvt_bf16_kernel<<<1024, 256, 0, stream>>>(base_w, wb);   // 1M elems
    prep_At_kernel <<< 512, 256, 0, stream>>>(lora_A, At);
    prep_Bt_kernel <<< 512, 256, 0, stream>>>(lora_B, Bt);
    router_kernel  <<< 512, 256, 0, stream>>>(x, router_w, router_b, ew);
    moe_ax_kernel  <<<  32, 256, 0, stream>>>(xb, At, ew, wax);
    dim3 grid(ODIM / 128, NTOK / 128);            // 8 x 32 tiles
    moe_main_kernel<<<grid, 256, 0, stream>>>(xb, wb, base_b, wax, Bt, out);
}
